// gat_ib_29102698398299
// MI455X (gfx1250) — compile-verified, matrix operand order checked by reading, not run on hardware
//
#include <hip/hip_runtime.h>
#include <hip/hip_bf16.h>
#include <cstddef>

// ---------------------------------------------------------------------------
// Problem constants (match reference)
// ---------------------------------------------------------------------------
#define NNODES 20000
#define NEDGES 640000
#define FIN1   128
#define FOUT   256      // H * C
#define NHEAD  4
#define CHEAD  64
#define NEG_SLOPE 0.2f

typedef __attribute__((ext_vector_type(16))) __bf16 v16bf;
typedef __attribute__((ext_vector_type(8)))  __bf16 v8bf;
typedef __attribute__((ext_vector_type(8)))  float  v8f;

static __device__ __forceinline__ unsigned fkey(float f) {
    unsigned i = __float_as_uint(f);
    return (i & 0x80000000u) ? ~i : (i | 0x80000000u);
}
static __device__ __forceinline__ float funkey(unsigned k) {
    unsigned i = (k & 0x80000000u) ? (k & 0x7fffffffu) : ~k;
    return __uint_as_float(i);
}

// ---------------------------------------------------------------------------
// f32 -> bf16 convert (for the activation matrix A, kept row-major)
// ---------------------------------------------------------------------------
__global__ void k_f32_to_bf16(const float* __restrict__ in, __bf16* __restrict__ out, long n) {
    long t = (long)blockIdx.x * blockDim.x + threadIdx.x;
    if (t < n) out[t] = (__bf16)in[t];
}

// ---------------------------------------------------------------------------
// Pack W[K,Nout] (f32, row-major) into WMMA B-fragment order (bf16):
//   Bp[(((tN*(K/32) + kblk)*32 + lane)*16 + i]
// where lane holds column n = tN*16 + (lane&15) and
//   K indices: kb=(lane>>4)*8;  i<8 -> kblk*32+kb+i ; i>=8 -> kblk*32+kb+8+i
// This turns the per-iteration B load in the GEMM into two b128 loads.
// ---------------------------------------------------------------------------
__global__ void k_pack_w(const float* __restrict__ W, __bf16* __restrict__ Bp,
                         int K, int Nout) {
    int t = blockIdx.x * blockDim.x + threadIdx.x;
    if (t >= K * Nout) return;
    const int i    = t & 15;
    const int lane = (t >> 4) & 31;
    const int rest = t >> 9;              // tN*(K/32) + kblk
    const int kblks = K >> 5;
    const int tN   = rest / kblks;
    const int kblk = rest - tN * kblks;
    const int n  = tN * 16 + (lane & 15);
    const int kb = (lane >> 4) << 3;
    const int k  = kblk * 32 + kb + ((i < 8) ? i : (i + 8));
    Bp[t] = (__bf16)W[(size_t)k * Nout + n];
}

// ---------------------------------------------------------------------------
// WMMA GEMM: C[M,Nout] = A[M,K] * B[K,Nout], bf16 in, f32 out.
// One 16x16 output tile per wave32; K templated so the loop fully unrolls
// (4 WMMAs for K=128, 8 for K=256). A row-major; B pre-packed in fragment
// order. All fragment traffic is b128 vector loads.
// ---------------------------------------------------------------------------
template <int K>
__global__ void k_gemm_bf16(const __bf16* __restrict__ A, const __bf16* __restrict__ Bp,
                            float* __restrict__ C, int M, int Nout) {
    const int lane   = threadIdx.x & 31;
    const int wid    = threadIdx.x >> 5;
    const int tilesN = Nout >> 4;
    const int wave   = blockIdx.x * (blockDim.x >> 5) + wid;
    const int tM = wave / tilesN;
    const int tN = wave - tM * tilesN;
    if (tM * 16 >= M) return;

    const int m  = lane & 15;            // A row within tile
    const int n  = lane & 15;            // C column within tile
    const int hi = lane >> 4;            // half-wave select
    const int kb = hi << 3;              // K sub-base: 0 or 8

    v8f acc = {};
    const __bf16* Arow  = A + (size_t)(tM * 16 + m) * K;
    const __bf16* Bfrag = Bp + ((size_t)tN * (K / 32) * 32 + lane) * 16;

#pragma unroll
    for (int kblk = 0; kblk < K / 32; ++kblk) {
        const int k0 = kblk * 32;
        // A fragment: lane holds row m, K = {k0+kb..+7, k0+kb+16..+23}
        v8bf alo = *(const v8bf*)(Arow + k0 + kb);
        v8bf ahi = *(const v8bf*)(Arow + k0 + kb + 16);
        // B fragment: contiguous 32B per lane (pre-packed)
        const __bf16* bptr = Bfrag + (size_t)kblk * 32 * 16;
        v8bf blo = *(const v8bf*)(bptr);
        v8bf bhi = *(const v8bf*)(bptr + 8);
        v16bf af, bfr;
#pragma unroll
        for (int i = 0; i < 8; ++i) {
            af[i] = alo[i];  af[8 + i]  = ahi[i];
            bfr[i] = blo[i]; bfr[8 + i] = bhi[i];
        }
        acc = __builtin_amdgcn_wmma_f32_16x16x32_bf16(
            false, af, false, bfr, (short)0, acc, false, false);
    }

    // D layout: VGPR r -> row r (lanes 0-15) / row r+8 (lanes 16-31), col = n
    const int rbase = hi * 8;
#pragma unroll
    for (int r = 0; r < 8; ++r)
        C[(size_t)(tM * 16 + rbase + r) * Nout + tN * 16 + n] = acc[r];
}

// ---------------------------------------------------------------------------
// ce[h] = dot(W_e[0, h*64 : h*64+64], a_e[h, :])   (edge_attr is scalar/edge)
// ---------------------------------------------------------------------------
__global__ void k_ce(const float* __restrict__ We, const float* __restrict__ ae,
                     float* __restrict__ ce) {
    __shared__ float red[256];
    int t = threadIdx.x, h = t >> 6, c = t & 63;
    red[t] = We[h * 64 + c] * ae[h * 64 + c];
    __syncthreads();
    for (int s = 32; s >= 1; s >>= 1) {
        if (c < s) red[t] += red[t + s];
        __syncthreads();
    }
    if (c == 0) ce[h] = red[t];
}

// ---------------------------------------------------------------------------
// Per-node attention logits: asrc[n,h], adst[n,h]
// ---------------------------------------------------------------------------
__global__ void k_node_alpha(const float* __restrict__ xh,
                             const float* __restrict__ a_src, const float* __restrict__ a_dst,
                             float* __restrict__ asrc, float* __restrict__ adst) {
    int t = blockIdx.x * blockDim.x + threadIdx.x;
    if (t >= NNODES * NHEAD) return;
    int nidx = t >> 2, h = t & 3;
    const float* row = xh + (size_t)nidx * FOUT + h * CHEAD;
    float s0 = 0.f, s1 = 0.f;
#pragma unroll 4
    for (int c = 0; c < CHEAD; ++c) {
        float v = row[c];
        s0 += v * a_src[h * CHEAD + c];
        s1 += v * a_dst[h * CHEAD + c];
    }
    asrc[t] = s0;
    adst[t] = s1;
}

// ---------------------------------------------------------------------------
// init segment-max keys (-inf key) and denominators
// ---------------------------------------------------------------------------
__global__ void k_init_seg(unsigned* __restrict__ segkey, float* __restrict__ denom, int n) {
    int t = blockIdx.x * blockDim.x + threadIdx.x;
    if (t < n) { segkey[t] = 0x007FFFFFu; denom[t] = 0.f; }
}

__global__ void k_zero(float* __restrict__ p, long n) {
    long t = (long)blockIdx.x * blockDim.x + threadIdx.x;
    if (t < n) p[t] = 0.f;
}

// ---------------------------------------------------------------------------
// Edge pass 1: logits + leaky-relu + segment max (monotone-uint atomicMax)
// ---------------------------------------------------------------------------
__global__ void k_edge1(const int* __restrict__ src, const int* __restrict__ dst,
                        const float* __restrict__ asrc, const float* __restrict__ adst,
                        const float* __restrict__ ce, const float* __restrict__ edge_attr,
                        float* __restrict__ wout, unsigned* __restrict__ segkey) {
    int t = blockIdx.x * blockDim.x + threadIdx.x;
    if (t >= NEDGES * NHEAD) return;
    int e = t >> 2, h = t & 3;
    int s = src[e], d = dst[e];
    float a = asrc[s * NHEAD + h] + adst[d * NHEAD + h] + ce[h] * edge_attr[e];
    a = (a > 0.f) ? a : NEG_SLOPE * a;
    wout[t] = a;
    atomicMax(&segkey[d * NHEAD + h], fkey(a));
}

// ---------------------------------------------------------------------------
// Edge pass 2: exp(a - max) + segment sum
// ---------------------------------------------------------------------------
__global__ void k_edge2(const int* __restrict__ dst,
                        float* __restrict__ wout, const unsigned* __restrict__ segkey,
                        float* __restrict__ denom) {
    int t = blockIdx.x * blockDim.x + threadIdx.x;
    if (t >= NEDGES * NHEAD) return;
    int e = t >> 2, h = t & 3;
    int d = dst[e];
    float v = __expf(wout[t] - funkey(segkey[d * NHEAD + h]));
    wout[t] = v;
    atomicAdd(&denom[d * NHEAD + h], v);
}

// ---------------------------------------------------------------------------
// Edge pass 3: normalize weights (written back, = returned attention) and
// scatter messages. One block per edge; 256 lanes = all (h,c) channels,
// coalesced gather from xh[src] (L2-resident) and coalesced f32 atomics to
// out[dst]. This is the throughput-limiting phase (~327M L2 atomics total).
// ---------------------------------------------------------------------------
__global__ void k_edge_scatter(const int* __restrict__ src, const int* __restrict__ dst,
                               const float* __restrict__ xh, float* __restrict__ wout,
                               const float* __restrict__ denom,
                               const float* __restrict__ edge_atten,
                               float* __restrict__ out) {
    int e = blockIdx.x;
    int c = threadIdx.x;           // 0..255
    int h = c >> 6;
    int s = src[e], d = dst[e];
    float wexp = wout[e * NHEAD + h];
    float den  = denom[d * NHEAD + h];
    float wn   = wexp / (den + 1e-16f);
    __syncthreads();               // all reads of wout done before rewrite
    if ((c & 63) == 0) wout[e * NHEAD + h] = wn;   // normalized alpha output
    float msg = xh[(size_t)s * FOUT + c] * wn * edge_atten[e];
    atomicAdd(&out[(size_t)d * FOUT + c], msg);
}

__global__ void k_bias(float* __restrict__ p, const float* __restrict__ b, long n) {
    long t = (long)blockIdx.x * blockDim.x + threadIdx.x;
    if (t < n) p[t] += b[t & (FOUT - 1)];
}

// ---------------------------------------------------------------------------
// Host side
// ---------------------------------------------------------------------------
static inline int cdiv(long a, int b) { return (int)((a + b - 1) / b); }

template <int K>
static void run_layer(const float* X,
                      const float* W, const float* a_src, const float* a_dst,
                      const float* W_e, const float* a_e, const float* bias,
                      const int* src, const int* dst,
                      const float* edge_attr, const float* edge_atten,
                      float* xh, __bf16* xb, __bf16* wb,
                      float* asrc, float* adst, unsigned* segkey, float* denom, float* ce,
                      float* outbuf, float* wout, hipStream_t stream) {
    // converts: A row-major bf16; W packed into fragment order
    k_f32_to_bf16<<<cdiv((long)NNODES * K, 256), 256, 0, stream>>>(X, xb, (long)NNODES * K);
    k_pack_w<<<cdiv((long)K * FOUT, 256), 256, 0, stream>>>(W, wb, K, FOUT);
    // GEMM: 1250*16 tiles, 4 waves / block of 128
    {
        int tiles = (NNODES / 16) * (FOUT / 16);
        k_gemm_bf16<K><<<tiles / 4, 128, 0, stream>>>(xb, wb, xh, NNODES, FOUT);
    }
    k_ce<<<1, 256, 0, stream>>>(W_e, a_e, ce);
    k_node_alpha<<<cdiv((long)NNODES * NHEAD, 256), 256, 0, stream>>>(xh, a_src, a_dst, asrc, adst);
    k_init_seg<<<cdiv((long)NNODES * NHEAD, 256), 256, 0, stream>>>(segkey, denom, NNODES * NHEAD);
    k_zero<<<cdiv((long)NNODES * FOUT, 256), 256, 0, stream>>>(outbuf, (long)NNODES * FOUT);
    k_edge1<<<cdiv((long)NEDGES * NHEAD, 256), 256, 0, stream>>>(src, dst, asrc, adst, ce, edge_attr,
                                                                 wout, segkey);
    k_edge2<<<cdiv((long)NEDGES * NHEAD, 256), 256, 0, stream>>>(dst, wout, segkey, denom);
    k_edge_scatter<<<NEDGES, 256, 0, stream>>>(src, dst, xh, wout, denom, edge_atten, outbuf);
    k_bias<<<cdiv((long)NNODES * FOUT, 256), 256, 0, stream>>>(outbuf, bias, (long)NNODES * FOUT);
}

extern "C" void kernel_launch(void* const* d_in, const int* in_sizes, int n_in,
                              void* d_out, int out_size, void* d_ws, size_t ws_size,
                              hipStream_t stream) {
    const float* x          = (const float*)d_in[0];
    const int*   ei         = (const int*)d_in[1];
    const float* edge_attr  = (const float*)d_in[3];
    const float* edge_atten = (const float*)d_in[4];
    const float* W1 = (const float*)d_in[5];
    const float* a_src1 = (const float*)d_in[6];
    const float* a_dst1 = (const float*)d_in[7];
    const float* W_e1 = (const float*)d_in[8];
    const float* a_e1 = (const float*)d_in[9];
    const float* b1 = (const float*)d_in[10];
    const float* W2 = (const float*)d_in[11];
    const float* a_src2 = (const float*)d_in[12];
    const float* a_dst2 = (const float*)d_in[13];
    const float* W_e2 = (const float*)d_in[14];
    const float* a_e2 = (const float*)d_in[15];
    const float* b2 = (const float*)d_in[16];

    const int* src = ei;
    const int* dst = ei + NEDGES;

    float* out_h  = (float*)d_out;                       // [N, 256]
    float* out_w1 = out_h + (size_t)NNODES * FOUT;       // [E, 4]
    float* out_w2 = out_w1 + (size_t)NEDGES * NHEAD;     // [E, 4]

    // workspace carve (total ~52.6 MB)
    char* w = (char*)d_ws;
    float*    xh     = (float*)w;    w += (size_t)NNODES * FOUT * 4;
    float*    hbuf   = (float*)w;    w += (size_t)NNODES * FOUT * 4;
    __bf16*   xb     = (__bf16*)w;   w += (size_t)NNODES * FOUT * 2;
    __bf16*   wb     = (__bf16*)w;   w += (size_t)FOUT * FOUT * 2;
    float*    asrc   = (float*)w;    w += (size_t)NNODES * NHEAD * 4;
    float*    adst   = (float*)w;    w += (size_t)NNODES * NHEAD * 4;
    unsigned* segkey = (unsigned*)w; w += (size_t)NNODES * NHEAD * 4;
    float*    denom  = (float*)w;    w += (size_t)NNODES * NHEAD * 4;
    float*    ce     = (float*)w;    w += 64;

    // Layer 1: x[N,128] -> hbuf[N,256]; normalized attention -> out_w1
    run_layer<FIN1>(x, W1, a_src1, a_dst1, W_e1, a_e1, b1,
                    src, dst, edge_attr, edge_atten,
                    xh, xb, wb, asrc, adst, segkey, denom, ce,
                    hbuf, out_w1, stream);

    // Layer 2: hbuf[N,256] -> out_h[N,256]; normalized attention -> out_w2
    run_layer<FOUT>(hbuf, W2, a_src2, a_dst2, W_e2, a_e2, b2,
                    src, dst, edge_attr, edge_atten,
                    xh, xb, wb, asrc, adst, segkey, denom, ce,
                    out_h, out_w2, stream);
}
